// AE_2714419331613
// MI455X (gfx1250) — compile-verified, matrix operand order checked by reading, not run on hardware
//
#include <hip/hip_runtime.h>

// ---------------------------------------------------------------------------
// MI455X (gfx1250) implementation. wave32, WMMA 16x16x32 f16 -> f32,
// TDM (tensor_load_to_lds) staging of f16 weight tiles.
// ---------------------------------------------------------------------------

typedef __attribute__((ext_vector_type(16))) _Float16 v16h;
typedef __attribute__((ext_vector_type(8)))  float    v8f;
typedef __attribute__((ext_vector_type(4)))  unsigned int v4u;
typedef __attribute__((ext_vector_type(8)))  int      v8i;
typedef __attribute__((ext_vector_type(4)))  int      v4i;

#define WMMA_F16(a, b, c) \
  __builtin_amdgcn_wmma_f32_16x16x32_f16(false, (a), false, (b), (short)0, (c), false, false)

// fragment K index map for 16x32 f16 A/B fragments (ISA 7.12.2):
// lane half (lane>>4) selects K group of 8; elems 0..7 -> K kg+0..7, 8..15 -> K 16+kg+0..7
__device__ __forceinline__ int frag_k(int j, int kg) {
  return (j < 8) ? (kg + j) : (16 + kg + (j - 8));
}

// ---------------------------------------------------------------------------
// f32 -> f16 convert (weights, done once; ~5 MB total)
// ---------------------------------------------------------------------------
__global__ void convert_f16_kernel(const float* __restrict__ x, _Float16* __restrict__ y,
                                   long long n) {
  long long i = (long long)blockIdx.x * blockDim.x + threadIdx.x;
  if (i < n) y[i] = (_Float16)x[i];
}

// ---------------------------------------------------------------------------
// Generic GEMM:  C[M,N] = act(A[M,K](f32) * W[K,N](f16) + bias)
// block = 256 threads (8 waves), tile 128x64, K-step 32.
// W tile (32x64 f16) staged LDS via Tensor Data Mover; A tile staged+converted
// by the threads. OT = float or _Float16.
// ---------------------------------------------------------------------------
template <typename OT, int ACT>
__global__ __launch_bounds__(256) void gemm_wmma_kernel(
    const float* __restrict__ A, const _Float16* __restrict__ W,
    const float* __restrict__ bias, OT* __restrict__ C, int M, int N, int K) {
  __shared__ _Float16 As[128][40];   // padded to dodge bank conflicts
  __shared__ _Float16 Bs[32 * 64];   // TDM writes tile rows contiguously

  const int tid   = threadIdx.x;
  const int wave  = tid >> 5;
  const int lane  = tid & 31;
  const int waveM = wave & 3;   // 4 waves along M
  const int waveN = wave >> 2;  // 2 waves along N
  const int blockM = blockIdx.y * 128;
  const int blockN = blockIdx.x * 64;
  const int mhalf = lane >> 4;
  const int ml    = lane & 15;
  const int kg    = mhalf * 8;

  v8f acc[2][2] = {};
  const int nK = (K + 31) / 32;

  for (int kt = 0; kt < nK; ++kt) {
    const int k0 = kt * 32;

    // ---- TDM: DMA the 32x64 f16 weight tile into LDS (wave 0 issues) ----
    if (wave == 0) {
      unsigned long long ga =
          (unsigned long long)(size_t)(W + (size_t)k0 * N + blockN);
      unsigned int ldsa = (unsigned int)(size_t)(&Bs[0]);  // flat->LDS offset (low 32 bits)
      v4u g0;
      g0[0] = 1u;                                           // count=1, user mode
      g0[1] = ldsa;                                         // lds_addr
      g0[2] = (unsigned int)(ga & 0xFFFFFFFFull);           // global_addr[31:0]
      g0[3] = (unsigned int)((ga >> 32) & 0x1FFFFFFull) | (2u << 30);  // ga[56:32] | type=2
      unsigned int td0 = (unsigned int)(N - blockN);        // remaining cols (OOB -> 0)
      unsigned int td1 = (unsigned int)(K - k0);            // remaining rows (OOB -> 0)
      unsigned long long st0 = (unsigned long long)N;       // row stride, data_size units
      v8i g1;
      g1[0] = (int)(1u << 16);                              // wg_mask=0, data_size=2B
      g1[1] = (int)((td0 & 0xFFFFu) << 16);                 // ab_addr=0 | tensor_dim0[15:0]
      g1[2] = (int)(((td0 >> 16) & 0xFFFFu) | ((td1 & 0xFFFFu) << 16));
      g1[3] = (int)(((td1 >> 16) & 0xFFFFu) | (64u << 16)); // tile_dim0 = 64
      g1[4] = (int)32u;                                     // tile_dim1 = 32, tile_dim2 = 0
      g1[5] = (int)(unsigned int)(st0 & 0xFFFFFFFFull);     // dim0_stride[31:0]
      g1[6] = (int)(unsigned int)((st0 >> 32) & 0xFFFFull); // dim0_stride[47:32]
      g1[7] = 0;
      v4i g2 = {0, 0, 0, 0};
      v4i g3 = {0, 0, 0, 0};
      v8i g4 = {0, 0, 0, 0, 0, 0, 0, 0};
      __builtin_amdgcn_tensor_load_to_lds(g0, g1, g2, g3, g4, 0);
      __builtin_amdgcn_s_wait_tensorcnt(0);
    }

    // ---- stage A tile 128x32 (f32 -> f16), 16 elems / thread ----
    #pragma unroll
    for (int i = 0; i < 16; ++i) {
      int idx = tid * 16 + i;
      int r = idx >> 5, c = idx & 31;
      int gr = blockM + r, gc = k0 + c;
      float v = 0.f;
      if (gr < M && gc < K) v = A[(long long)gr * K + gc];
      As[r][c] = (_Float16)v;
    }
    // prefetch next A tile (global_prefetch_b8)
    if (kt + 1 < nK) {
      int pr = blockM + (tid >> 1); if (pr >= M) pr = M - 1;
      int pc = k0 + 32;             if (pc >= K) pc = K - 1;
      __builtin_prefetch((const void*)(A + (long long)pr * K + pc), 0, 1);
    }
    __syncthreads();

    v16h af[2], bf[2];
    #pragma unroll
    for (int mt = 0; mt < 2; ++mt) {
      int row = waveM * 32 + mt * 16 + ml;
      #pragma unroll
      for (int j = 0; j < 16; ++j) af[mt][j] = As[row][frag_k(j, kg)];
    }
    #pragma unroll
    for (int nt = 0; nt < 2; ++nt) {
      int col = waveN * 32 + nt * 16 + ml;
      #pragma unroll
      for (int j = 0; j < 16; ++j) bf[nt][j] = Bs[frag_k(j, kg) * 64 + col];
    }
    #pragma unroll
    for (int mt = 0; mt < 2; ++mt)
      #pragma unroll
      for (int nt = 0; nt < 2; ++nt)
        acc[mt][nt] = WMMA_F16(af[mt], bf[nt], acc[mt][nt]);
    __syncthreads();
  }

  // ---- epilogue: bias (+ReLU), store (C layout: vgpr r -> row r + 8*mhalf) ----
  #pragma unroll
  for (int mt = 0; mt < 2; ++mt)
    #pragma unroll
    for (int nt = 0; nt < 2; ++nt) {
      int gcol = blockN + waveN * 32 + nt * 16 + ml;
      float bv = (gcol < N) ? bias[gcol] : 0.f;
      #pragma unroll
      for (int r = 0; r < 8; ++r) {
        int grow = blockM + waveM * 32 + mt * 16 + r + 8 * mhalf;
        if (grow < M && gcol < N) {
          float v = acc[mt][nt][r] + bv;
          if (ACT == 1) v = v > 0.f ? v : 0.f;
          C[(long long)grow * N + gcol] = (OT)v;
        }
      }
    }
}

// ---------------------------------------------------------------------------
// Fused flash attention for the co-attention blocks (nh=4, dh=128).
// Q/K/V are f16 [B, L, 512]; one wave per (b, h, 16-query tile).
// scores = Q Kt via WMMA, streaming softmax, ctx = P V via WMMA
// (P converted C-layout -> A-layout through LDS).
// ---------------------------------------------------------------------------
__global__ __launch_bounds__(32) void attn_wmma_kernel(
    const _Float16* __restrict__ Q, const _Float16* __restrict__ Km,
    const _Float16* __restrict__ Vm, float* __restrict__ O,
    int Lq, int Lk, float scale) {
  const int DHT = 512;  // nh*dh
  const int qt = blockIdx.x, h = blockIdx.y, b = blockIdx.z;
  const int lane = threadIdx.x;
  const int half = lane >> 4, nl = lane & 15, kg = half * 8;
  const int q0 = qt * 16;

  const long long baseQ = (long long)b * Lq * DHT + h * 128;
  const long long baseK = (long long)b * Lk * DHT + h * 128;

  // Q fragments: 4 K-chunks of 32 covering dh=128
  int qrow = q0 + nl; if (qrow >= Lq) qrow = Lq - 1;
  const _Float16* qp = Q + baseQ + (long long)qrow * DHT;
  v16h qf[4];
  #pragma unroll
  for (int c = 0; c < 4; ++c)
    #pragma unroll
    for (int j = 0; j < 16; ++j) qf[c][j] = qp[c * 32 + frag_k(j, kg)];

  v8f o[8] = {};
  float mrow[8], lrow[8];
  #pragma unroll
  for (int r = 0; r < 8; ++r) { mrow[r] = -1e30f; lrow[r] = 0.f; }

  __shared__ _Float16 Pl[16][40];

  for (int kt = 0; kt < Lk; kt += 32) {
    // ---- S = Q * K^T for a 16x32 key tile (two 16x16 WMMA C tiles) ----
    v8f s[2] = {};
    #pragma unroll
    for (int nt = 0; nt < 2; ++nt) {
      int key = kt + nt * 16 + nl;
      int keyc = key < Lk ? key : Lk - 1;
      const _Float16* kp = Km + baseK + (long long)keyc * DHT;
      #pragma unroll
      for (int c = 0; c < 4; ++c) {
        v16h bf;
        #pragma unroll
        for (int j = 0; j < 16; ++j) bf[j] = kp[c * 32 + frag_k(j, kg)];
        s[nt] = WMMA_F16(qf[c], bf, s[nt]);
      }
    }
    // scale + key-padding mask (mask depends only on this lane's column)
    #pragma unroll
    for (int nt = 0; nt < 2; ++nt) {
      int key = kt + nt * 16 + nl;
      #pragma unroll
      for (int r = 0; r < 8; ++r) {
        float v = s[nt][r] * scale;
        s[nt][r] = (key < Lk) ? v : -1e30f;
      }
    }
    // ---- streaming softmax: per-row max over 32 cols (shfl over 16 lanes) ----
    float corr[8];
    #pragma unroll
    for (int r = 0; r < 8; ++r) {
      float t = fmaxf(s[0][r], s[1][r]);
      #pragma unroll
      for (int off = 1; off < 16; off <<= 1) t = fmaxf(t, __shfl_xor(t, off, 32));
      float nm = fmaxf(mrow[r], t);
      corr[r] = __expf(mrow[r] - nm);
      mrow[r] = nm;
    }
    #pragma unroll
    for (int r = 0; r < 8; ++r) {
      float p0 = __expf(s[0][r] - mrow[r]);
      float p1 = __expf(s[1][r] - mrow[r]);
      s[0][r] = p0; s[1][r] = p1;
      float t = p0 + p1;
      #pragma unroll
      for (int off = 1; off < 16; off <<= 1) t += __shfl_xor(t, off, 32);
      lrow[r] = lrow[r] * corr[r] + t;
    }
    #pragma unroll
    for (int dt = 0; dt < 8; ++dt)
      #pragma unroll
      for (int r = 0; r < 8; ++r) o[dt][r] *= corr[r];

    // ---- P: C layout -> A layout through LDS ----
    __syncthreads();
    #pragma unroll
    for (int r = 0; r < 8; ++r) {
      Pl[r + 8 * half][nl]      = (_Float16)s[0][r];
      Pl[r + 8 * half][16 + nl] = (_Float16)s[1][r];
    }
    __syncthreads();
    v16h pf;
    #pragma unroll
    for (int j = 0; j < 16; ++j) pf[j] = Pl[nl][frag_k(j, kg)];

    // ---- ctx += P(16x32) * V(32x128) ----
    #pragma unroll
    for (int dt = 0; dt < 8; ++dt) {
      v16h vf;
      #pragma unroll
      for (int j = 0; j < 16; ++j) {
        int key = kt + frag_k(j, kg);
        if (key >= Lk) key = Lk - 1;  // masked keys carry p ~ 0
        vf[j] = Vm[baseK + (long long)key * DHT + dt * 16 + nl];
      }
      o[dt] = WMMA_F16(pf, vf, o[dt]);
    }
  }

  // ---- normalize + store ----
  #pragma unroll
  for (int dt = 0; dt < 8; ++dt)
    #pragma unroll
    for (int r = 0; r < 8; ++r) {
      int grow = q0 + r + 8 * half;
      if (grow < Lq)
        O[((long long)b * Lq + grow) * DHT + h * 128 + dt * 16 + nl] = o[dt][r] / lrow[r];
    }
}

// ---------------------------------------------------------------------------
// Small kernels
// ---------------------------------------------------------------------------
__global__ __launch_bounds__(256) void ln_res_kernel(
    const float* __restrict__ X, const float* __restrict__ R,
    const float* __restrict__ g, const float* __restrict__ bta,
    float* __restrict__ Y, int M) {
  int wave = threadIdx.x >> 5, lane = threadIdx.x & 31;
  int row = blockIdx.x * 8 + wave;
  if (row >= M) return;
  const float* x = X + (long long)row * 128;
  const float* r = R + (long long)row * 128;
  float v[4]; float s = 0.f;
  #pragma unroll
  for (int i = 0; i < 4; ++i) { v[i] = x[lane + 32 * i] + r[lane + 32 * i]; s += v[i]; }
  #pragma unroll
  for (int off = 16; off > 0; off >>= 1) s += __shfl_xor(s, off, 32);
  float mean = s * (1.f / 128.f);
  float ss = 0.f;
  #pragma unroll
  for (int i = 0; i < 4; ++i) { float d = v[i] - mean; ss += d * d; }
  #pragma unroll
  for (int off = 16; off > 0; off >>= 1) ss += __shfl_xor(ss, off, 32);
  float inv = rsqrtf(ss * (1.f / 128.f) + 1e-5f);
  #pragma unroll
  for (int i = 0; i < 4; ++i)
    Y[(long long)row * 128 + lane + 32 * i] =
        (v[i] - mean) * inv * g[lane + 32 * i] + bta[lane + 32 * i];
}

__global__ void mean_rows_kernel(const float* __restrict__ X, const float* __restrict__ addv,
                                 float* __restrict__ Y, int L, float alpha) {
  int b = blockIdx.x, d = threadIdx.x;  // 128 threads
  float s = 0.f;
  const float* x = X + (long long)b * L * 128 + d;
  for (int l = 0; l < L; ++l) s += x[(long long)l * 128];
  s *= alpha;
  if (addv) s += addv[b * 128 + d];
  Y[b * 128 + d] = s;
}

// 2-token transformer MHA (nh=2, dh=64); qkv = [M=2B, 384] (q|k|v slices)
__global__ void attn2_small_kernel(const float* __restrict__ qkv, float* __restrict__ ctx) {
  int t = blockIdx.x * blockDim.x + threadIdx.x;
  if (t >= 256 * 2 * 2) return;
  int b = t >> 2, h = (t >> 1) & 1, qi = t & 1;
  const float* q  = qkv + (long long)(b * 2 + qi) * 384 + h * 64;
  const float* k0 = qkv + (long long)(b * 2 + 0) * 384 + 128 + h * 64;
  const float* k1 = qkv + (long long)(b * 2 + 1) * 384 + 128 + h * 64;
  const float* v0 = qkv + (long long)(b * 2 + 0) * 384 + 256 + h * 64;
  const float* v1 = qkv + (long long)(b * 2 + 1) * 384 + 256 + h * 64;
  float s0 = 0.f, s1 = 0.f;
  for (int d = 0; d < 64; ++d) { s0 += q[d] * k0[d]; s1 += q[d] * k1[d]; }
  s0 *= 0.125f; s1 *= 0.125f;  // 64^-0.5
  float m = fmaxf(s0, s1);
  float e0 = __expf(s0 - m), e1 = __expf(s1 - m);
  float inv = 1.f / (e0 + e1);
  e0 *= inv; e1 *= inv;
  float* o = ctx + (long long)(b * 2 + qi) * 128 + h * 64;
  for (int d = 0; d < 64; ++d) o[d] = e0 * v0[d] + e1 * v1[d];
}

__device__ __forceinline__ float block_sum_128(float x, float* sbuf, int d) {
  sbuf[d] = x; __syncthreads();
  for (int off = 64; off > 0; off >>= 1) {
    if (d < off) sbuf[d] += sbuf[d + off];
    __syncthreads();
  }
  float r = sbuf[0]; __syncthreads();
  return r;
}

// fused = ((v + (1-cos(v,c))(c-v)) + (a + (1-cos(a,h))(h-a)) + t) / 3
__global__ __launch_bounds__(128) void fuse_adjust_kernel(
    const float* __restrict__ vision, const float* __restrict__ clip,
    const float* __restrict__ audio, const float* __restrict__ hub,
    const float* __restrict__ text, float* __restrict__ fused) {
  __shared__ float sbuf[128];
  int b = blockIdx.x, d = threadIdx.x;
  float vp = vision[b * 128 + d], cp = clip[b * 128 + d];
  float ap = audio[b * 128 + d],  hp = hub[b * 128 + d];
  float tp = text[b * 128 + d];
  float dvc = block_sum_128(vp * cp, sbuf, d);
  float nv  = sqrtf(block_sum_128(vp * vp, sbuf, d));
  float nc  = sqrtf(block_sum_128(cp * cp, sbuf, d));
  float dah = block_sum_128(ap * hp, sbuf, d);
  float na  = sqrtf(block_sum_128(ap * ap, sbuf, d));
  float nh  = sqrtf(block_sum_128(hp * hp, sbuf, d));
  float simv = dvc / (fmaxf(nv, 1e-8f) * fmaxf(nc, 1e-8f));
  float sima = dah / (fmaxf(na, 1e-8f) * fmaxf(nh, 1e-8f));
  float vadj = vp + (1.f - simv) * (cp - vp);
  float aadj = ap + (1.f - sima) * (hp - ap);
  fused[b * 128 + d] = (vadj + aadj + tp) * (1.f / 3.f);
}

__global__ void sub_kernel(const float* __restrict__ a, const float* __restrict__ b,
                           float* __restrict__ y, int n) {
  int i = blockIdx.x * blockDim.x + threadIdx.x;
  if (i < n) y[i] = a[i] - b[i];
}

__global__ void concat3_kernel(const float* __restrict__ a, const float* __restrict__ b,
                               const float* __restrict__ c, float* __restrict__ y) {
  int t = blockIdx.x * blockDim.x + threadIdx.x;  // 256*384
  if (t >= 256 * 384) return;
  int bb = t / 384, j = t % 384;
  float v = (j < 128) ? a[bb * 128 + j]
          : (j < 256) ? b[bb * 128 + (j - 128)]
                      : c[bb * 128 + (j - 256)];
  y[t] = v;
}

__global__ void interleave2_kernel(const float* __restrict__ a, const float* __restrict__ b,
                                   float* __restrict__ y) {
  int t = blockIdx.x * blockDim.x + threadIdx.x;  // 256*128
  if (t >= 256 * 128) return;
  int bb = t / 128, d = t % 128;
  y[(long long)(2 * bb) * 128 + d]     = a[t];
  y[(long long)(2 * bb + 1) * 128 + d] = b[t];
}

// ---------------------------------------------------------------------------
// Host-side wiring
// ---------------------------------------------------------------------------
template <int ACT>
static void launch_gemm_f32(const float* A, const _Float16* W, const float* bias, float* C,
                            int M, int N, int K, hipStream_t s) {
  dim3 g((N + 63) / 64, (M + 127) / 128);
  gemm_wmma_kernel<float, ACT><<<g, 256, 0, s>>>(A, W, bias, C, M, N, K);
}
static void launch_gemm_f16(const float* A, const _Float16* W, const float* bias, _Float16* C,
                            int M, int N, int K, hipStream_t s) {
  dim3 g((N + 63) / 64, (M + 127) / 128);
  gemm_wmma_kernel<_Float16, 0><<<g, 256, 0, s>>>(A, W, bias, C, M, N, K);
}

// param leaf indices (setup_inputs dict insertion order, depth-first)
enum {
  IN_SEM = 0, IN_EMO, IN_VIS, IN_AUD,
  P_TE_W = 4, P_TE_B, P_TS_W, P_TS_B, P_IMG_W, P_IMG_B, P_AU_W, P_AU_B,
  TE_INW = 12, TE_INB, TE_OW, TE_OB, TE_G1, TE_B1, TE_F1W, TE_F1B, TE_F2W, TE_F2B, TE_G2, TE_B2,
  TS_INW = 24, TS_INB, TS_OW, TS_OB, TS_G1, TS_B1, TS_F1W, TS_F1B, TS_F2W, TS_F2B, TS_G2, TS_B2,
  CA_LVW = 36, CA_LVB, CA_LSW, CA_LSB,
  MV_QW = 40, MV_QB, MV_KW, MV_KB, MV_VW, MV_VB, MV_PW, MV_PB, MV_G, MV_B,
  MS_QW = 50, MS_QB, MS_KW, MS_KB, MS_VW, MS_VB, MS_PW, MS_PB, MS_G, MS_B,
  DN_1W = 60, DN_1B, DN_2W, DN_2B,
  CL_1W = 64, CL_1B, CL_2W, CL_2B,
  IN_NF = 68, IN_NR = 69
};

struct TrmP {
  const _Float16 *Wq, *Wo, *Wf1, *Wf2;
  const float *bq, *bo, *g1, *b1, *bf1, *bf2, *g2, *b2;
};

struct TrmScratch { float *qkv, *ctx2, *sa, *x1, *hbuf, *ff, *x2; };

static void run_trm(const float* x0, const TrmP& p, const TrmScratch& t,
                    float* mean_out, hipStream_t s) {
  const int M = 512;  // 2 tokens * B
  launch_gemm_f32<0>(x0, p.Wq, p.bq, t.qkv, M, 384, 128, s);
  attn2_small_kernel<<<(1024 + 127) / 128, 128, 0, s>>>(t.qkv, t.ctx2);
  launch_gemm_f32<0>(t.ctx2, p.Wo, p.bo, t.sa, M, 128, 128, s);
  ln_res_kernel<<<(M + 7) / 8, 256, 0, s>>>(x0, t.sa, p.g1, p.b1, t.x1, M);
  launch_gemm_f32<1>(t.x1, p.Wf1, p.bf1, t.hbuf, M, 2048, 128, s);
  launch_gemm_f32<0>(t.hbuf, p.Wf2, p.bf2, t.ff, M, 128, 2048, s);
  ln_res_kernel<<<(M + 7) / 8, 256, 0, s>>>(t.x1, t.ff, p.g2, p.b2, t.x2, M);
  mean_rows_kernel<<<256, 128, 0, s>>>(t.x2, nullptr, mean_out, 2, 0.5f);
}

extern "C" void kernel_launch(void* const* d_in, const int* in_sizes, int n_in,
                              void* d_out, int out_size, void* d_ws, size_t ws_size,
                              hipStream_t stream) {
  (void)in_sizes; (void)n_in; (void)out_size; (void)ws_size;
  const int B = 256, LS = 512, LV = 83;
  const int MS = B * LS;   // 131072
  const int MV = B * LV;   // 21248
  const float SC_BIG = 0.08838834764831845f;  // 128^-0.5

  auto PF = [&](int i) { return (const float*)d_in[i]; };

  size_t off = 0;
  auto alloc = [&](size_t bytes) -> void* {
    void* p = (char*)d_ws + off;
    off += (bytes + 255) & ~(size_t)255;
    return p;
  };
  auto f32buf = [&](size_t n) { return (float*)alloc(n * 4); };
  auto f16buf = [&](size_t n) { return (_Float16*)alloc(n * 2); };

  // ---- convert all GEMM weights to f16 once (TDM stages f16 tiles) ----
  auto cvtW = [&](int idx, size_t n) -> const _Float16* {
    _Float16* p = f16buf(n);
    convert_f16_kernel<<<(unsigned)((n + 255) / 256), 256, 0, stream>>>(PF(idx), p, (long long)n);
    return p;
  };
  const _Float16* wTE  = cvtW(P_TE_W, 768 * 128);
  const _Float16* wAU  = cvtW(P_AU_W, 768 * 128);
  const _Float16* wTS  = cvtW(P_TS_W, 512 * 128);
  const _Float16* wIMG = cvtW(P_IMG_W, 512 * 128);
  const _Float16* wTEin = cvtW(TE_INW, 128 * 384);
  const _Float16* wTEo  = cvtW(TE_OW, 128 * 128);
  const _Float16* wTEf1 = cvtW(TE_F1W, 128 * 2048);
  const _Float16* wTEf2 = cvtW(TE_F2W, 2048 * 128);
  const _Float16* wTSin = cvtW(TS_INW, 128 * 384);
  const _Float16* wTSo  = cvtW(TS_OW, 128 * 128);
  const _Float16* wTSf1 = cvtW(TS_F1W, 128 * 2048);
  const _Float16* wTSf2 = cvtW(TS_F2W, 2048 * 128);
  const _Float16* wLV  = cvtW(CA_LVW, 128 * 128);
  const _Float16* wLSm = cvtW(CA_LSW, 128 * 128);
  const _Float16* wMVq = cvtW(MV_QW, 128 * 512);
  const _Float16* wMVk = cvtW(MV_KW, 128 * 512);
  const _Float16* wMVv = cvtW(MV_VW, 128 * 512);
  const _Float16* wMVp = cvtW(MV_PW, 512 * 128);
  const _Float16* wMSq = cvtW(MS_QW, 128 * 512);
  const _Float16* wMSk = cvtW(MS_KW, 128 * 512);
  const _Float16* wMSv = cvtW(MS_VW, 128 * 512);
  const _Float16* wMSp = cvtW(MS_PW, 512 * 128);
  const _Float16* wDN1 = cvtW(DN_1W, 128 * 512);
  const _Float16* wDN2 = cvtW(DN_2W, 512 * 128);
  const _Float16* wCL1 = cvtW(CL_1W, 384 * 128);
  const _Float16* wCL2 = cvtW(CL_2W, 128 * 2);

  // ---- workspace ----
  float* t_emo  = f32buf((size_t)B * 128);
  float* a_emo  = f32buf((size_t)B * 128);
  float* t_sem  = f32buf((size_t)MS * 128);
  float* v_fea  = f32buf((size_t)MV * 128);
  float* x0e    = f32buf((size_t)512 * 128);
  float* x0s    = f32buf((size_t)512 * 128);
  TrmScratch ts;
  ts.qkv  = f32buf((size_t)512 * 384);
  ts.ctx2 = f32buf((size_t)512 * 128);
  ts.sa   = f32buf((size_t)512 * 128);
  ts.x1   = f32buf((size_t)512 * 128);
  ts.hbuf = f32buf((size_t)512 * 2048);
  ts.ff   = f32buf((size_t)512 * 128);
  ts.x2   = f32buf((size_t)512 * 128);
  float* fusion_emo = f32buf((size_t)B * 128);
  float* fusion_sem = f32buf((size_t)B * 128);
  float* text_pool  = f32buf((size_t)B * 128);
  float* vis_pool   = f32buf((size_t)B * 128);
  float* v_    = f32buf((size_t)MV * 128);
  float* s_    = f32buf((size_t)MS * 128);
  _Float16* q_v  = f16buf((size_t)MV * 512);
  _Float16* k_v  = f16buf((size_t)MS * 512);
  _Float16* vv_v = f16buf((size_t)MS * 512);
  float* ctx_v = f32buf((size_t)MV * 512);
  float* pr_v  = f32buf((size_t)MV * 128);
  float* cont_v = f32buf((size_t)MV * 128);
  _Float16* q_s  = f16buf((size_t)MS * 512);
  _Float16* k_s  = f16buf((size_t)MV * 512);
  _Float16* v_s  = f16buf((size_t)MV * 512);
  float* ctx_s = f32buf((size_t)MS * 512);
  float* pr_s  = f32buf((size_t)MS * 128);
  float* cont_t = f32buf((size_t)MS * 128);
  float* clip_tg = f32buf((size_t)B * 128);  // == cv
  float* ct_mean = f32buf((size_t)B * 128);
  float* fused   = f32buf((size_t)B * 128);
  float* den1    = f32buf((size_t)B * 512);
  float* den2    = f32buf((size_t)B * 128);
  float* fdiff   = f32buf((size_t)B * 128);
  float* AE      = f32buf((size_t)B * 384);
  float* c1      = f32buf((size_t)B * 128);

  // ---- 1. modality MLPs (Linear + ReLU) ----
  launch_gemm_f32<1>(PF(IN_EMO), wTE, PF(P_TE_B), t_emo, B, 128, 768, stream);
  launch_gemm_f32<1>(PF(IN_AUD), wAU, PF(P_AU_B), a_emo, B, 128, 768, stream);
  launch_gemm_f32<1>(PF(IN_SEM), wTS, PF(P_TS_B), t_sem, MS, 128, 512, stream);
  launch_gemm_f32<1>(PF(IN_VIS), wIMG, PF(P_IMG_B), v_fea, MV, 128, 512, stream);

  // ---- 2. emotion transformer (2 tokens) -> fusion_emo ----
  interleave2_kernel<<<(B * 128 + 255) / 256, 256, 0, stream>>>(t_emo, a_emo, x0e);
  TrmP pe = {wTEin, wTEo, wTEf1, wTEf2,
             PF(TE_INB), PF(TE_OB), PF(TE_G1), PF(TE_B1),
             PF(TE_F1B), PF(TE_F2B), PF(TE_G2), PF(TE_B2)};
  run_trm(x0e, pe, ts, fusion_emo, stream);

  // ---- 3. pooled features ----
  mean_rows_kernel<<<B, 128, 0, stream>>>(t_sem, t_emo, text_pool, LS, 1.f / LS);
  mean_rows_kernel<<<B, 128, 0, stream>>>(v_fea, nullptr, vis_pool, LV, 1.f / LV);

  // ---- 4. co-attention ----
  launch_gemm_f32<0>(v_fea, wLV, PF(CA_LVB), v_, MV, 128, 128, stream);
  launch_gemm_f32<0>(t_sem, wLSm, PF(CA_LSB), s_, MS, 128, 128, stream);
  // mha_v: queries = vision, keys/values = text
  launch_gemm_f16(v_, wMVq, PF(MV_QB), q_v, MV, 512, 128, stream);
  launch_gemm_f16(s_, wMVk, PF(MV_KB), k_v, MS, 512, 128, stream);
  launch_gemm_f16(s_, wMVv, PF(MV_VB), vv_v, MS, 512, 128, stream);
  attn_wmma_kernel<<<dim3((LV + 15) / 16, 4, B), 32, 0, stream>>>(
      q_v, k_v, vv_v, ctx_v, LV, LS, SC_BIG);
  launch_gemm_f32<0>(ctx_v, wMVp, PF(MV_PB), pr_v, MV, 128, 512, stream);
  ln_res_kernel<<<(MV + 7) / 8, 256, 0, stream>>>(v_, pr_v, PF(MV_G), PF(MV_B), cont_v, MV);
  // mha_s: queries = text, keys/values = vision
  launch_gemm_f16(s_, wMSq, PF(MS_QB), q_s, MS, 512, 128, stream);
  launch_gemm_f16(v_, wMSk, PF(MS_KB), k_s, MV, 512, 128, stream);
  launch_gemm_f16(v_, wMSv, PF(MS_VB), v_s, MV, 512, 128, stream);
  attn_wmma_kernel<<<dim3(LS / 16, 4, B), 32, 0, stream>>>(
      q_s, k_s, v_s, ctx_s, LS, LV, SC_BIG);
  launch_gemm_f32<0>(ctx_s, wMSp, PF(MS_PB), pr_s, MS, 128, 512, stream);
  ln_res_kernel<<<(MS + 7) / 8, 256, 0, stream>>>(s_, pr_s, PF(MS_G), PF(MS_B), cont_t, MS);

  // ---- 5. targets / pools from co-attention ----
  mean_rows_kernel<<<B, 128, 0, stream>>>(cont_v, nullptr, clip_tg, LV, 1.f / LV);  // clip == cv
  mean_rows_kernel<<<B, 128, 0, stream>>>(cont_t, nullptr, ct_mean, LS, 1.f / LS);

  // ---- 6. cosine adjust + denoise - noise_rev ----
  fuse_adjust_kernel<<<B, 128, 0, stream>>>(vis_pool, clip_tg, a_emo, fusion_emo,
                                            text_pool, fused);
  launch_gemm_f32<1>(fused, wDN1, PF(DN_1B), den1, B, 512, 128, stream);
  launch_gemm_f32<0>(den1, wDN2, PF(DN_2B), den2, B, 128, 512, stream);
  sub_kernel<<<(B * 128 + 255) / 256, 256, 0, stream>>>(den2, PF(IN_NR), fdiff, B * 128);

  // ---- 7. semantic transformer on [ct, cv] -> fusion_sem ----
  interleave2_kernel<<<(B * 128 + 255) / 256, 256, 0, stream>>>(ct_mean, clip_tg, x0s);
  TrmP psm = {wTSin, wTSo, wTSf1, wTSf2,
              PF(TS_INB), PF(TS_OB), PF(TS_G1), PF(TS_B1),
              PF(TS_F1B), PF(TS_F2B), PF(TS_G2), PF(TS_B2)};
  run_trm(x0s, psm, ts, fusion_sem, stream);

  // ---- 8. classifier ----
  concat3_kernel<<<(B * 384 + 255) / 256, 256, 0, stream>>>(fusion_emo, fusion_sem, fdiff, AE);
  launch_gemm_f32<1>(AE, wCL1, PF(CL_1B), c1, B, 128, 384, stream);
  launch_gemm_f32<0>(c1, wCL2, PF(CL_2B), (float*)d_out, B, 2, 128, stream);
}
